// CSPNGenerate_37056977830362
// MI455X (gfx1250) — matrix operand order repaired and checked by reading, not compiled, not run on hardware
//
#include <hip/hip_runtime.h>

typedef __attribute__((ext_vector_type(16))) _Float16 v16h;
typedef __attribute__((ext_vector_type(8)))  _Float16 v8h;
typedef __attribute__((ext_vector_type(8)))  float    v8f;

constexpr int NB     = 4;     // batch
constexpr int CIN    = 64;    // input channels
constexpr int OC     = 8;     // conv output channels
constexpr int H      = 352;
constexpr int W      = 1216;
constexpr int HP     = 354;   // H + 2
constexpr int WP     = 1218;  // W + 2
constexpr int TILE_W = 64;    // pixels per block (4 waves x 16)
constexpr int WIN_W  = 66;    // TILE_W + halo
constexpr int NCHUNK = 18;    // K = 9 taps * 64 ch = 576 -> 18 chunks of 32
constexpr int WFRAG_ELEMS = NCHUNK * 32 * 16;   // 9216 halves = 18 KB
constexpr int FEAT_ELEMS  = 3 * WIN_W * CIN;    // 12672

__global__ void zero_out_kernel(float4* __restrict__ out, int n4) {
    int i = blockIdx.x * blockDim.x + threadIdx.x;
    if (i < n4) out[i] = float4{0.f, 0.f, 0.f, 0.f};
}

// One-shot: swizzle conv weights into the exact WMMA B-fragment layout.
// K ordering: k = t*64 + c (t = kh*3+kw). Chunk kc covers c in [(kc&1)*32, +32).
// B order used here (contiguous k, same as A): lane l -> column n = l&15; element e -> K = e + 16*(l>>4).
__global__ void swizzle_w_kernel(const float* __restrict__ wgt,
                                 _Float16* __restrict__ wfrag) {
    int idx = blockIdx.x * blockDim.x + threadIdx.x;
    if (idx >= WFRAG_ELEMS) return;
    int e  = idx & 15;
    int l  = (idx >> 4) & 31;
    int kc = idx >> 9;
    int n  = l & 15;
    int kk = e + 16 * (l >> 4);              // 0..31 within chunk
    int c  = ((kc & 1) << 5) + kk;
    int t  = kc >> 1;                        // tap index 0..8
    _Float16 v = (_Float16)0.0f;
    if (n < OC) v = (_Float16)wgt[((n * CIN + c) * 9) + t];
    wfrag[idx] = v;
}

template <bool USE_WFRAG>
__global__ __launch_bounds__(128) void cspn_guide_wmma_kernel(
    const float* __restrict__ feat,          // (NB, CIN, H, W)
    const float* __restrict__ wgt,           // (OC, CIN, 3, 3)  (fallback path)
    const _Float16* __restrict__ wfrag,      // pre-swizzled B fragments in ws
    const float* __restrict__ gamma,
    const float* __restrict__ beta,
    const float* __restrict__ mean,
    const float* __restrict__ var,
    float* __restrict__ out)                 // (NB, 9, HP, WP), pre-zeroed
{
    // feature window, channel-last: [row 0..2][col 0..65][c 0..63], f16
    __shared__ __align__(16) _Float16 sFeat[3 * WIN_W * CIN];

    const int tid  = threadIdx.x;
    const int lane = tid & 31;
    const int wave = tid >> 5;                // 0..3
    const int wx0  = blockIdx.x * TILE_W;     // tile origin in W
    const int h    = blockIdx.y;
    const int b    = blockIdx.z;

    // ---- Stage feature window rows h-1..h+1, cols wx0-1..wx0+64 (zero-padded) ----
    // idx = rc*WIN_W + cl, cl fastest -> consecutive x per consecutive thread (coalesced).
    // Incremental div/mod by 66 (step 128 = 66 + 62) avoids integer division.
    {
        int cl = (tid < WIN_W) ? tid : tid - WIN_W;
        int rc = (tid < WIN_W) ? 0 : 1;
        for (int idx = tid; idx < FEAT_ELEMS; idx += 128) {
            int c   = rc & (CIN - 1);
            int row = rc >> 6;
            int y = h + row - 1;
            int x = wx0 + cl - 1;
            float v = 0.0f;
            if ((unsigned)y < (unsigned)H && (unsigned)x < (unsigned)W)
                v = feat[(((size_t)b * CIN + c) * H + y) * W + x];
            sFeat[(row * WIN_W + cl) * CIN + c] = (_Float16)v;
            cl += 62; rc += 1;
            if (cl >= WIN_W) { cl -= WIN_W; rc += 1; }
        }
    }

    // ---- B-fragment source: global workspace (L2/WGP$-resident) or LDS fallback ----
    const _Float16* bbase;
    if constexpr (USE_WFRAG) {
        bbase = wfrag;
    } else {
        __shared__ __align__(16) _Float16 sW[WFRAG_ELEMS];
        for (int idx = tid; idx < WFRAG_ELEMS; idx += 128) {
            int e  = idx & 15;
            int l  = (idx >> 4) & 31;
            int kc = idx >> 9;
            int n  = l & 15;
            int kk = e + 16 * (l >> 4);
            int c  = ((kc & 1) << 5) + kk;
            int t  = kc >> 1;
            _Float16 v = (_Float16)0.0f;
            if (n < OC) v = (_Float16)wgt[((n * CIN + c) * 9) + t];
            sW[idx] = v;
        }
        bbase = sW;
    }

    __syncthreads();

    // ---- Implicit GEMM: per wave M=16 pixels, N=16 (8 OC), K=576 ----
    const int woff = wave * 16;       // this wave's pixel offset in tile
    const int m    = lane & 15;       // pixel within wave tile
    const int hi   = lane >> 4;       // lane half-group
    v8f acc = {};

    const _Float16* blane = bbase + lane * 16;

    for (int kc = 0; kc < NCHUNK; ++kc) {
        int t  = kc >> 1;
        int kh = t / 3, kw = t % 3;
        int c0 = (kc & 1) << 5;
        int cl = woff + m + kw;       // window column for pixel + tap

        // A k order used here (contiguous, not the ISA table order): lanes 0-15 hold K 0-15, lanes 16-31 hold K 16-31
        // (element e -> K = e + 16*hi, matching B). Channel-last LDS -> contiguous runs.
        const _Float16* ap = &sFeat[(kh * WIN_W + cl) * CIN + c0 + 16 * hi];
        v8h a0 = *(const v8h*)(ap);
        v8h a1 = *(const v8h*)(ap + 8);

        const _Float16* bp = blane + kc * (32 * 16);
        v8h b0 = *(const v8h*)(bp);
        v8h b1 = *(const v8h*)(bp + 8);

        v16h A, Bf;
        #pragma unroll
        for (int i = 0; i < 8; ++i) {
            A[i] = a0[i]; A[i + 8] = a1[i];
            Bf[i] = b0[i]; Bf[i + 8] = b1[i];
        }
        acc = __builtin_amdgcn_wmma_f32_16x16x32_f16(
            /*neg_a=*/false, A, /*neg_b=*/false, Bf,
            /*c_mod=*/(short)0, acc, /*reuse_a=*/false, /*reuse_b=*/false);
    }

    __syncthreads();   // all waves done reading sFeat -> reuse it as float stage

    // C/D layout: VGPR r -> M = r + 8*hi, lane -> N = lane&15.
    float* stage = (float*)sFeat;     // [64 pixels][16 cols], 4 KB < 25.3 KB
    #pragma unroll
    for (int r = 0; r < 8; ++r)
        stage[(woff + r + 8 * hi) * 16 + m] = acc[r];

    __syncthreads();

    // ---- Epilogue: BN -> L1 normalize -> mid channel -> scatter into padded out ----
    if (tid < TILE_W) {
        int p = tid;
        int x = wx0 + p;
        if (x < W) {
            float g[OC];
            float asum = 0.0f;
            #pragma unroll
            for (int n = 0; n < OC; ++n) {
                float sc = gamma[n] * rsqrtf(var[n] + 1e-5f);
                float sh = beta[n] - mean[n] * sc;
                float v  = stage[p * 16 + n] * sc + sh;
                g[n] = v;
                asum += fabsf(v);
            }
            float inv = 1.0f / asum;
            float ssum = 0.0f;
            #pragma unroll
            for (int n = 0; n < OC; ++n) { g[n] *= inv; ssum += g[n]; }
            float mid = 1.0f - ssum;

            #pragma unroll
            for (int t = 0; t < 9; ++t) {
                float v = (t < 4) ? g[t] : ((t == 4) ? mid : g[t - 1]);
                int oy = h + t / 3;
                int ox = x + t % 3;
                out[(((size_t)b * 9 + t) * HP + oy) * WP + ox] = v;
            }
        }
    }
}

extern "C" void kernel_launch(void* const* d_in, const int* in_sizes, int n_in,
                              void* d_out, int out_size, void* d_ws, size_t ws_size,
                              hipStream_t stream) {
    const float* feat  = (const float*)d_in[0];
    const float* wgt   = (const float*)d_in[1];
    const float* gamma = (const float*)d_in[2];
    const float* beta  = (const float*)d_in[3];
    const float* mean  = (const float*)d_in[4];
    const float* var   = (const float*)d_in[5];
    float* out = (float*)d_out;

    // borders of the padded output must be zero (out_size divisible by 4)
    int n4 = out_size >> 2;
    zero_out_kernel<<<(n4 + 255) / 256, 256, 0, stream>>>((float4*)out, n4);

    dim3 grid(W / TILE_W, H, NB);   // 19 x 352 x 4
    if (ws_size >= (size_t)WFRAG_ELEMS * sizeof(_Float16)) {
        _Float16* wfrag = (_Float16*)d_ws;
        swizzle_w_kernel<<<(WFRAG_ELEMS + 127) / 128, 128, 0, stream>>>(wgt, wfrag);
        cspn_guide_wmma_kernel<true><<<grid, 128, 0, stream>>>(
            feat, wgt, wfrag, gamma, beta, mean, var, out);
    } else {
        cspn_guide_wmma_kernel<false><<<grid, 128, 0, stream>>>(
            feat, wgt, nullptr, gamma, beta, mean, var, out);
    }
}
